// MultiHeadAttention_86552180949460
// MI455X (gfx1250) — matrix operand order repaired and checked by reading, not compiled, not run on hardware
//
#include <hip/hip_runtime.h>

// ---------------------------------------------------------------------------
// MI455X (gfx1250) multi-head attention: f16 WMMA pipeline, f32 accumulate.
//   k1: weights fp32 -> f16
//   k2: fused QKV projection GEMM (head-split outputs, V transposed)
//   k3: flash attention (online softmax, LDS transpose for P)
//   k4: output projection GEMM -> fp32
// ---------------------------------------------------------------------------

typedef __attribute__((ext_vector_type(16))) _Float16 v16h;
typedef __attribute__((ext_vector_type(8)))  _Float16 v8h;
typedef __attribute__((ext_vector_type(8)))  float    v8f;
typedef __attribute__((ext_vector_type(4)))  float    v4f;

#define D_MODEL 768
#define NHEADS  12
#define DK      64
#define BATCH   2
#define SEQ     2048
#define ROWS    (BATCH * SEQ)               // 4096
#define WELEMS  (D_MODEL * D_MODEL)         // 589824
#define HELEMS  (BATCH * NHEADS * SEQ * DK) // 3145728

static __device__ inline v8f wmma_f16(v16h a, v16h b, v8f c) {
  // D = A(16x32 f16) * B(32x16 f16) + C(16x16 f32)
  return __builtin_amdgcn_wmma_f32_16x16x32_f16(
      /*neg_a=*/false, a, /*neg_b=*/false, b,
      /*c_mod=*/(short)0, c, /*reuse_a=*/false, /*reuse_b=*/false);
}

static __device__ inline v16h joinh(v8h lo, v8h hi) {
  v16h r;
#pragma unroll
  for (int i = 0; i < 8; ++i) { r[i] = lo[i]; r[i + 8] = hi[i]; }
  return r;
}

// f16 A-operand (16x32) slice for this lane.
// A layout: lane l -> m = l&15, g = l>>4; element e: k = g*16 + e
static __device__ inline v16h load_a_f16(const _Float16* row_ptr, int g) {
  v8h lo = *(const v8h*)(row_ptr + g * 16);
  v8h hi = *(const v8h*)(row_ptr + g * 16 + 8);
  return joinh(lo, hi);
}

// fp32 A-operand slice with on-the-fly f16 conversion (same k mapping).
static __device__ inline v16h load_a_f32(const float* row_ptr, int g) {
  v4f f0 = *(const v4f*)(row_ptr + g * 16);
  v4f f1 = *(const v4f*)(row_ptr + g * 16 + 4);
  v4f f2 = *(const v4f*)(row_ptr + g * 16 + 8);
  v4f f3 = *(const v4f*)(row_ptr + g * 16 + 12);
  v16h r;
#pragma unroll
  for (int i = 0; i < 4; ++i) {
    r[i]      = (_Float16)f0[i];
    r[4 + i]  = (_Float16)f1[i];
    r[8 + i]  = (_Float16)f2[i];
    r[12 + i] = (_Float16)f3[i];
  }
  return r;
}

// B layout (32x16): lane l -> n = l&15, g = l>>4; element e: k = g*16 + e
static __device__ inline v16h load_b_f16(const _Float16* p) {
  return joinh(*(const v8h*)p, *(const v8h*)(p + 8));
}

// ---------------------------------------------------------------------------
// Kernel 1: convert the four fp32 weight matrices to f16 (contiguous in ws).
// ---------------------------------------------------------------------------
__global__ __launch_bounds__(256) void cvtw_kernel(
    const float* __restrict__ wq, const float* __restrict__ wk,
    const float* __restrict__ wv, const float* __restrict__ wo,
    _Float16* __restrict__ dst) {
  int i = blockIdx.x * 256 + threadIdx.x;
  if (i < WELEMS) {
    dst[i]              = (_Float16)wq[i];
    dst[WELEMS + i]     = (_Float16)wk[i];
    dst[2 * WELEMS + i] = (_Float16)wv[i];
    dst[3 * WELEMS + i] = (_Float16)wo[i];
  }
}

// ---------------------------------------------------------------------------
// Kernel 2: fused QKV projection. Y = X @ W.T (W row-major [out,in] is the
// natural WMMA B operand). Head-split f16 outputs:
//   mode 0/1: Qh/Kh  [B,H,S,64]
//   mode 2:   Vt     [B,H,64,S]   (transposed so PV B-operand is contiguous)
// WG: 256 threads = 8 waves (2x4); wave tile 32x64 (2x4 WMMA tiles);
// WG tile 64(M) x 256(N). Grid = (4096/64, 768/256, 3).
// ---------------------------------------------------------------------------
__global__ __launch_bounds__(256) void proj_kernel(
    const float* __restrict__ qx, const float* __restrict__ kx,
    const float* __restrict__ vx,
    const _Float16* __restrict__ Wq, const _Float16* __restrict__ Wk,
    const _Float16* __restrict__ Wv,
    _Float16* __restrict__ Qh, _Float16* __restrict__ Kh,
    _Float16* __restrict__ Vt) {
  const int mode = blockIdx.z;
  const float*    X = (mode == 0) ? qx : (mode == 1) ? kx : vx;
  const _Float16* W = (mode == 0) ? Wq : (mode == 1) ? Wk : Wv;

  const int lane = threadIdx.x & 31;
  const int wave = threadIdx.x >> 5;
  const int g  = lane >> 4;
  const int ln = lane & 15;
  const int wm = wave >> 2;  // 0..1
  const int wn = wave & 3;   // 0..3
  const int M0 = blockIdx.x * 64 + wm * 32;
  const int N0 = blockIdx.y * 256 + wn * 64;

  v8f acc[2][4] = {};

  for (int kb = 0; kb < D_MODEL; kb += 32) {
    v16h a[2], b[4];
#pragma unroll
    for (int mt = 0; mt < 2; ++mt) {
      const float* xp = X + (size_t)(M0 + mt * 16 + ln) * D_MODEL + kb;
      a[mt] = load_a_f32(xp, g);
    }
#pragma unroll
    for (int nt = 0; nt < 4; ++nt) {
      const _Float16* wp =
          W + (size_t)(N0 + nt * 16 + ln) * D_MODEL + kb + g * 16;
      b[nt] = load_b_f16(wp);
    }
#pragma unroll
    for (int mt = 0; mt < 2; ++mt)
#pragma unroll
      for (int nt = 0; nt < 4; ++nt)
        acc[mt][nt] = wmma_f16(a[mt], b[nt], acc[mt][nt]);
  }

  // Scatter-store f16 with head permute. C/D layout: m = g*8+r, n = ln.
  if (mode == 2) {
#pragma unroll
    for (int mt = 0; mt < 2; ++mt)
#pragma unroll
      for (int nt = 0; nt < 4; ++nt)
#pragma unroll
        for (int r = 0; r < 8; ++r) {
          int mg = M0 + mt * 16 + g * 8 + r;  // 0..4095
          int ng = N0 + nt * 16 + ln;         // 0..767
          int bb = mg >> 11, s = mg & (SEQ - 1);
          int h = ng >> 6, dk = ng & 63;
          size_t head = (size_t)(bb * NHEADS + h);
          Vt[(head * DK + dk) * SEQ + s] = (_Float16)acc[mt][nt][r];
        }
  } else {
    _Float16* dst = (mode == 0) ? Qh : Kh;
#pragma unroll
    for (int mt = 0; mt < 2; ++mt)
#pragma unroll
      for (int nt = 0; nt < 4; ++nt)
#pragma unroll
        for (int r = 0; r < 8; ++r) {
          int mg = M0 + mt * 16 + g * 8 + r;
          int ng = N0 + nt * 16 + ln;
          int bb = mg >> 11, s = mg & (SEQ - 1);
          int h = ng >> 6, dk = ng & 63;
          size_t head = (size_t)(bb * NHEADS + h);
          dst[(head * SEQ + s) * DK + dk] = (_Float16)acc[mt][nt][r];
        }
  }
}

// ---------------------------------------------------------------------------
// Kernel 3: flash attention. One wave owns 16 query rows; 8 waves = 128 rows
// per WG. Key blocks of 64 (16 WMMAs per block). Grid = (S/128, H, B).
// ---------------------------------------------------------------------------
__global__ __launch_bounds__(256) void attn_kernel(
    const _Float16* __restrict__ Qh, const _Float16* __restrict__ Kh,
    const _Float16* __restrict__ Vt, _Float16* __restrict__ O16) {
  __shared__ alignas(16) _Float16 lds[8 * 16 * 64];  // 16 KB: P transpose slabs

  const int lane = threadIdx.x & 31;
  const int wave = threadIdx.x >> 5;
  const int g  = lane >> 4;
  const int ln = lane & 15;
  const int h  = blockIdx.y;
  const int b  = blockIdx.z;
  const size_t head = (size_t)(b * NHEADS + h);

  const _Float16* Q = Qh + head * SEQ * DK;
  const _Float16* K = Kh + head * SEQ * DK;
  const _Float16* V = Vt + head * DK * SEQ;

  const int q0 = blockIdx.x * 128 + wave * 16;
  _Float16* lp = lds + wave * (16 * 64);  // this wave's 16x64 slab

  // Q A-operands for the two 32-wide k-dim steps (dk 0..31, 32..63), resident
  // in registers for the whole key loop.
  v16h qa[2];
#pragma unroll
  for (int kk = 0; kk < 2; ++kk)
    qa[kk] = load_a_f16(Q + (size_t)(q0 + ln) * DK + kk * 32, g);

  v8f o[4] = {};  // 16x64 output accumulator (4 dk-tiles)
  float mrow[8], lrow[8];
#pragma unroll
  for (int r = 0; r < 8; ++r) { mrow[r] = -3.0e38f; lrow[r] = 0.0f; }

  const float scale = 0.125f;  // 1/sqrt(64)

  for (int kb = 0; kb < SEQ; kb += 64) {
    // ---- scores: S = Q @ K^T for 64 keys -> four 16x16 f32 tiles ----
    v8f c[4] = {};
#pragma unroll
    for (int nt = 0; nt < 4; ++nt) {
#pragma unroll
      for (int kk = 0; kk < 2; ++kk) {
        const _Float16* kp =
            K + (size_t)(kb + nt * 16 + ln) * DK + kk * 32 + g * 16;
        c[nt] = wmma_f16(qa[kk], load_b_f16(kp), c[nt]);
      }
    }

    // ---- online softmax (rows m = g*8+r live across lanes of one half) ----
    float alpha[8];
#pragma unroll
    for (int r = 0; r < 8; ++r) {
      float s0 = (float)c[0][r] * scale;
      float s1 = (float)c[1][r] * scale;
      float s2 = (float)c[2][r] * scale;
      float s3 = (float)c[3][r] * scale;
      float mx = fmaxf(fmaxf(s0, s1), fmaxf(s2, s3));
      mx = fmaxf(mx, __shfl_xor(mx, 1));
      mx = fmaxf(mx, __shfl_xor(mx, 2));
      mx = fmaxf(mx, __shfl_xor(mx, 4));
      mx = fmaxf(mx, __shfl_xor(mx, 8));
      float mnew = fmaxf(mrow[r], mx);
      float p0 = __expf(s0 - mnew);
      float p1 = __expf(s1 - mnew);
      float p2 = __expf(s2 - mnew);
      float p3 = __expf(s3 - mnew);
      float rs = (p0 + p1) + (p2 + p3);
      rs += __shfl_xor(rs, 1);
      rs += __shfl_xor(rs, 2);
      rs += __shfl_xor(rs, 4);
      rs += __shfl_xor(rs, 8);
      alpha[r] = __expf(mrow[r] - mnew);
      lrow[r] = lrow[r] * alpha[r] + rs;
      mrow[r] = mnew;
      // Stash P (f16) into LDS row-major [m][key_local] for the transpose.
      int m = g * 8 + r;
      lp[m * 64 + ln]      = (_Float16)p0;
      lp[m * 64 + 16 + ln] = (_Float16)p1;
      lp[m * 64 + 32 + ln] = (_Float16)p2;
      lp[m * 64 + 48 + ln] = (_Float16)p3;
    }

    // Rescale the running output accumulator.
#pragma unroll
    for (int dt = 0; dt < 4; ++dt)
#pragma unroll
      for (int r = 0; r < 8; ++r) o[dt][r] *= alpha[r];

    // Wave-local LDS store->load ordering (DS ops are in-order per wave; the
    // explicit wait also fences the compiler).
    asm volatile("s_wait_dscnt 0" ::: "memory");

    // Reload P as two A-operands (lane = query row, elements = key k-dim).
    v16h pa[2];
#pragma unroll
    for (int kk = 0; kk < 2; ++kk)
      pa[kk] = load_a_f16(lp + ln * 64 + kk * 32, g);

    // ---- PV: O += P(16x64) @ V(64x16) per dk tile ----
#pragma unroll
    for (int dt = 0; dt < 4; ++dt) {
#pragma unroll
      for (int kk = 0; kk < 2; ++kk) {
        const _Float16* vp =
            V + (size_t)(dt * 16 + ln) * SEQ + kb + kk * 32 + g * 16;
        o[dt] = wmma_f16(pa[kk], load_b_f16(vp), o[dt]);
      }
    }
  }

  // ---- normalize and store as f16 in [B,S,H*Dk] for the output GEMM ----
#pragma unroll
  for (int r = 0; r < 8; ++r) {
    float inv = 1.0f / lrow[r];
    int qrow = q0 + g * 8 + r;
    size_t base = ((size_t)(b * SEQ + qrow)) * D_MODEL + h * DK;
#pragma unroll
    for (int dt = 0; dt < 4; ++dt)
      O16[base + dt * 16 + ln] = (_Float16)(o[dt][r] * inv);
  }
}

// ---------------------------------------------------------------------------
// Kernel 4: output projection. out = O16(4096x768 f16) @ Wo16^T, fp32 store.
// Same tiling as proj: wave 32x64, WG 64x256. Grid = (64, 3).
// ---------------------------------------------------------------------------
__global__ __launch_bounds__(256) void oproj_kernel(
    const _Float16* __restrict__ O16, const _Float16* __restrict__ Wo,
    float* __restrict__ out) {
  const int lane = threadIdx.x & 31;
  const int wave = threadIdx.x >> 5;
  const int g  = lane >> 4;
  const int ln = lane & 15;
  const int wm = wave >> 2;
  const int wn = wave & 3;
  const int M0 = blockIdx.x * 64 + wm * 32;
  const int N0 = blockIdx.y * 256 + wn * 64;

  v8f acc[2][4] = {};

  for (int kb = 0; kb < D_MODEL; kb += 32) {
    v16h a[2], b[4];
#pragma unroll
    for (int mt = 0; mt < 2; ++mt) {
      const _Float16* ap = O16 + (size_t)(M0 + mt * 16 + ln) * D_MODEL + kb;
      a[mt] = load_a_f16(ap, g);
    }
#pragma unroll
    for (int nt = 0; nt < 4; ++nt) {
      const _Float16* wp =
          Wo + (size_t)(N0 + nt * 16 + ln) * D_MODEL + kb + g * 16;
      b[nt] = load_b_f16(wp);
    }
#pragma unroll
    for (int mt = 0; mt < 2; ++mt)
#pragma unroll
      for (int nt = 0; nt < 4; ++nt)
        acc[mt][nt] = wmma_f16(a[mt], b[nt], acc[mt][nt]);
  }

#pragma unroll
  for (int mt = 0; mt < 2; ++mt)
#pragma unroll
    for (int nt = 0; nt < 4; ++nt)
#pragma unroll
      for (int r = 0; r < 8; ++r) {
        int mg = M0 + mt * 16 + g * 8 + r;
        int ng = N0 + nt * 16 + ln;
        out[(size_t)mg * D_MODEL + ng] = acc[mt][nt][r];
      }
}

// ---------------------------------------------------------------------------
// Host launcher.
// ---------------------------------------------------------------------------
extern "C" void kernel_launch(void* const* d_in, const int* in_sizes, int n_in,
                              void* d_out, int out_size, void* d_ws,
                              size_t ws_size, hipStream_t stream) {
  (void)in_sizes; (void)n_in; (void)out_size; (void)ws_size;

  const float* q  = (const float*)d_in[0];
  const float* k  = (const float*)d_in[1];
  const float* v  = (const float*)d_in[2];
  const float* wq = (const float*)d_in[3];
  const float* wk = (const float*)d_in[4];
  const float* wv = (const float*)d_in[5];
  const float* wo = (const float*)d_in[6];
  float* out = (float*)d_out;

  _Float16* ws  = (_Float16*)d_ws;
  _Float16* W16 = ws;                       // 4 * WELEMS halves
  _Float16* Qh  = ws + 4 * (size_t)WELEMS;  // HELEMS halves each
  _Float16* Kh  = Qh + (size_t)HELEMS;
  _Float16* Vt  = Kh + (size_t)HELEMS;
  _Float16* O16 = Vt + (size_t)HELEMS;

  cvtw_kernel<<<(WELEMS + 255) / 256, 256, 0, stream>>>(wq, wk, wv, wo, W16);
  proj_kernel<<<dim3(ROWS / 64, D_MODEL / 256, 3), 256, 0, stream>>>(
      q, k, v, W16, W16 + WELEMS, W16 + 2 * (size_t)WELEMS, Qh, Kh, Vt);
  attn_kernel<<<dim3(SEQ / 128, NHEADS, BATCH), 256, 0, stream>>>(Qh, Kh, Vt,
                                                                  O16);
  oproj_kernel<<<dim3(ROWS / 64, D_MODEL / 256, 1), 256, 0, stream>>>(
      O16, W16 + 3 * (size_t)WELEMS, out);
}
